// QwenCausalSelfAttention_22265110463405
// MI455X (gfx1250) — compile-verified, matrix operand order checked by reading, not run on hardware
//
#include <hip/hip_runtime.h>
#include <hip/hip_bf16.h>

// ---------------------------------------------------------------------------
// CDNA5 (gfx1250, wave32) WMMA implementation of Qwen GQA causal attention.
// B=2, T=2048, DIM=2048, H=16, KVH=2, HD=128.  All matmuls via
// v_wmma_f32_16x16x32_bf16 (fp32 accumulate).  GEMMs stage via the async
// global->LDS DMA path (ASYNCcnt, double buffered); the attention kernel
// stages K/V tiles with the Tensor Data Mover (TENSORcnt).
// ---------------------------------------------------------------------------

#define Bc   2
#define Tc   2048
#define DIMc 2048
#define Hc   16
#define KVHc 2
#define HDc  128
#define Mc   (Bc * Tc)          // 4096 rows for all GEMMs
#define Kc   DIMc               // 2048 reduction dim for all GEMMs

typedef __attribute__((ext_vector_type(16))) __bf16 v16bf;
typedef __attribute__((ext_vector_type(8)))  float  v8f;
typedef __attribute__((ext_vector_type(4)))  unsigned tdm_u4;
typedef __attribute__((ext_vector_type(8)))  int      tdm_i8;
typedef __attribute__((ext_vector_type(4)))  int      tdm_i4;

__device__ __forceinline__ __bf16 f2bf(float f) {
  unsigned u = __builtin_bit_cast(unsigned, f);
  unsigned r = (u + 0x7FFFu + ((u >> 16) & 1u)) >> 16;   // round-to-nearest-even
  unsigned short s = (unsigned short)r;
  return __builtin_bit_cast(__bf16, s);
}

__device__ __forceinline__ v8f wmma_bf16(v16bf a, v16bf b, v8f c) {
  // 8-arg form: (neg_a, A, neg_b, B, c_mod, C, reuse_a, reuse_b)
  return __builtin_amdgcn_wmma_f32_16x16x32_bf16(false, a, false, b, (short)0, c,
                                                 false, false);
}

// --------------------- CDNA5 async global->LDS staging ---------------------
__device__ __forceinline__ void async_ld32(const void* g, unsigned ldsoff) {
  asm volatile(
      "global_load_async_to_lds_b128 %0, %1, off\n\t"
      "global_load_async_to_lds_b128 %0, %1, off offset:16"
      :: "v"(ldsoff), "v"(g) : "memory");
}
__device__ __forceinline__ void wait_async0() {
  asm volatile("s_wait_asynccnt 0x0" ::: "memory");
}
__device__ __forceinline__ unsigned lds_off(const void* p) {
  return (unsigned)(size_t)p;   // low 32 bits of generic ptr = LDS byte address
}

// --------------------------- CDNA5 Tensor Data Mover ------------------------
// 2D tile load, bf16 elements.  Descriptor per CDNA5 ISA:
//   group0: [1:0]=count=1 | lds_addr | global_addr[56:0] | type=2 (bits 127:126)
//   group1: data_size=1(2B) @ [17:16]; tensor_dim0 @ [79:48]; tensor_dim1 @
//           [111:80]; tile_dim0 @ [127:112]; tile_dim1 @ [143:128];
//           tensor_dim0_stride @ [207:160].  Groups 2/3 zero (2D tensor).
// LDS receives the tile packed row-major [tile_d1][tile_d0].
__device__ __forceinline__ void tdm_load_2d(const void* gaddr, unsigned ldsaddr,
                                            unsigned tensor_d0, unsigned tensor_d1,
                                            unsigned tile_d0, unsigned tile_d1,
                                            unsigned stride0) {
  unsigned long long ga = (unsigned long long)(size_t)gaddr;
  tdm_u4 g0;
  g0.x = 1u;                                            // count=1, user desc
  g0.y = ldsaddr;                                       // LDS byte address
  g0.z = (unsigned)ga;                                  // global addr [31:0]
  g0.w = (unsigned)((ga >> 32) & 0x01FFFFFFu) | (2u << 30);  // [56:32] | type=2
  tdm_i8 g1;
  g1[0] = (int)(1u << 16);                              // data_size = 2 bytes
  g1[1] = (int)((tensor_d0 & 0xFFFFu) << 16);           // tensor_dim0 lo16
  g1[2] = (int)(((tensor_d0 >> 16) & 0xFFFFu) |
                ((tensor_d1 & 0xFFFFu) << 16));         // dim0 hi | dim1 lo
  g1[3] = (int)(((tensor_d1 >> 16) & 0xFFFFu) |
                ((tile_d0 & 0xFFFFu) << 16));           // dim1 hi | tile_dim0
  g1[4] = (int)(tile_d1 & 0xFFFFu);                     // tile_dim1 (tile_dim2=0)
  g1[5] = (int)stride0;                                 // tensor_dim0_stride lo32
  g1[6] = 0;                                            // stride hi / dim1_stride
  g1[7] = 0;
  tdm_i4 gz = {0, 0, 0, 0};
#if defined(__clang_major__) && __clang_major__ >= 23
  tdm_i8 gz8 = {0, 0, 0, 0, 0, 0, 0, 0};
  __builtin_amdgcn_tensor_load_to_lds(g0, g1, gz, gz, gz8, 0);
#else
  __builtin_amdgcn_tensor_load_to_lds(g0, g1, gz, gz, 0);
#endif
}

// Load one 16x32 bf16 WMMA fragment (A-style; B-style identical when the
// tile is stored [n][k]).  ISA layout: lane = {half = lane>>4, row = lane&15};
// => two contiguous 16-byte groups at k offsets {half*8, 16+half*8}.
__device__ __forceinline__ v16bf ld_frag(const __bf16* base, int rstride,
                                         int kbase, int lane) {
  int h = (lane >> 4) & 1;
  int r = lane & 15;
  const __bf16* rp = base + (size_t)r * rstride + kbase + h * 8;
  union { v16bf v; uint4 q[2]; } f;
  f.q[0] = *(const uint4*)(rp);
  f.q[1] = *(const uint4*)(rp + 16);
  return f.v;
}

// ---------------------------------------------------------------------------
// Conversion kernels: fp32 -> bf16 (straight) and fp32 [K][N] -> bf16 [N][K].
// ---------------------------------------------------------------------------
__global__ __launch_bounds__(256) void cvt_f32_bf16(const float* __restrict__ s,
                                                    __bf16* __restrict__ d, int n) {
  int i = blockIdx.x * blockDim.x + threadIdx.x;
  int stride = gridDim.x * blockDim.x;
  for (; i < n; i += stride) d[i] = f2bf(s[i]);
}

__global__ __launch_bounds__(256) void tr_f32_bf16(const float* __restrict__ s,
                                                   __bf16* __restrict__ d,
                                                   int K, int N) {
  int i = blockIdx.x * blockDim.x + threadIdx.x;
  if (i >= K * N) return;
  int k = i / N, n = i % N;
  d[(size_t)n * K + k] = f2bf(s[i]);   // coalesced read, scattered write (one-time)
}

// ---------------------------------------------------------------------------
// Tiled GEMM:  C[M x N] = A[M x K](bf16) * Wt[N x K]^T(bf16)  (+bias, +RoPE)
// Block: 256 threads = 8 waves; tile 128(M) x 128(N); BK = 32.
// Double-buffered LDS, async-to-LDS staging overlapped with WMMA compute.
// LAYOUT: 0 = bf16 out [B,Hn,T,HD]   (Q, K)
//         1 = bf16 out [B,Hn,HD,T]   (V, pre-transposed for attention)
//         2 = f32  out row-major [M,N]  (final projection -> d_out)
// ---------------------------------------------------------------------------
template <int LAYOUT, bool ROPE, bool BIAS>
__global__ __launch_bounds__(256) void gemm_kernel(
    const __bf16* __restrict__ A, const __bf16* __restrict__ Wt,
    const float* __restrict__ bias, const float* __restrict__ cosb,
    const float* __restrict__ sinb, void* __restrict__ outp, int N, int Hn) {
  __shared__ __bf16 As[2][128][32];   // 2 x 8 KB
  __shared__ __bf16 Ws[2][128][32];   // 2 x 8 KB

  const int wave = threadIdx.x >> 5;
  const int lane = threadIdx.x & 31;
  const int m0 = blockIdx.y * 128;
  const int n0 = blockIdx.x * 128;

  v8f acc[8];
  const v8f vzero = {0.f, 0.f, 0.f, 0.f, 0.f, 0.f, 0.f, 0.f};
#pragma unroll
  for (int i = 0; i < 8; ++i) acc[i] = vzero;

  const int srow = threadIdx.x >> 1;          // 0..127
  const int sseg = (threadIdx.x & 1) * 16;    // 0 or 16
  const __bf16* agbase = A + (size_t)(m0 + srow) * Kc + sseg;
  const __bf16* wgbase = Wt + (size_t)(n0 + srow) * Kc + sseg;

  // prologue: stage k-tile 0 into buffer 0
  async_ld32(agbase, lds_off(&As[0][srow][sseg]));
  async_ld32(wgbase, lds_off(&Ws[0][srow][sseg]));
  wait_async0();
  __syncthreads();

  int buf = 0;
  for (int k0 = 0; k0 < Kc; k0 += 32) {
    if (k0 + 32 < Kc) {   // issue next tile's DMA; compute hides it
      async_ld32(agbase + k0 + 32, lds_off(&As[buf ^ 1][srow][sseg]));
      async_ld32(wgbase + k0 + 32, lds_off(&Ws[buf ^ 1][srow][sseg]));
    }
    v16bf afrag = ld_frag(&As[buf][wave * 16][0], 32, 0, lane);
    v16bf bfr[8];
#pragma unroll
    for (int ni = 0; ni < 8; ++ni)
      bfr[ni] = ld_frag(&Ws[buf][ni * 16][0], 32, 0, lane);
#pragma unroll
    for (int ni = 0; ni < 8; ++ni)
      acc[ni] = wmma_bf16(afrag, bfr[ni], acc[ni]);
    wait_async0();
    __syncthreads();
    buf ^= 1;
  }

  // ---------------- epilogue: bias, RoPE, store -----------------------------
  const int half = lane >> 4;
  const int ncol = lane & 15;

  if (BIAS) {
#pragma unroll
    for (int ni = 0; ni < 8; ++ni) {
      float bv = bias[n0 + (ni << 4) + ncol];
      acc[ni] += bv;
    }
  }

  if (ROPE) {
    // BN==128 == HEAD_DIM, so tiles (i, i+4) are the d / d+64 rotate pairs.
#pragma unroll
    for (int r = 0; r < 8; ++r) {
      int m = m0 + (wave << 4) + (half << 3) + r;
      int t = m & (Tc - 1);
#pragma unroll
      for (int i = 0; i < 4; ++i) {
        int dlo = (i << 4) + ncol;
        int dhi = dlo + 64;
        float clo = cosb[t * HDc + dlo], slo = sinb[t * HDc + dlo];
        float chi = cosb[t * HDc + dhi], shi = sinb[t * HDc + dhi];
        float a = acc[i][r];
        float b = acc[i + 4][r];
        acc[i][r]     = a * clo - b * slo;   // d <  64: q*c - q[d+64]*s
        acc[i + 4][r] = b * chi + a * shi;   // d >= 64: q*c + q[d-64]*s
      }
    }
  }

#pragma unroll
  for (int ni = 0; ni < 8; ++ni) {
#pragma unroll
    for (int r = 0; r < 8; ++r) {
      int m = m0 + (wave << 4) + (half << 3) + r;
      int bb = m >> 11;            // m / T   (T = 2048)
      int t  = m & (Tc - 1);       // m % T
      int d  = (ni << 4) + ncol;
      float v = acc[ni][r];
      if (LAYOUT == 0) {           // [B, Hn, T, HD]
        __bf16* ob = (__bf16*)outp;
        ob[(((size_t)(bb * Hn + blockIdx.x)) * Tc + t) * HDc + d] = f2bf(v);
      } else if (LAYOUT == 1) {    // [B, Hn, HD, T]  (V transposed)
        __bf16* ob = (__bf16*)outp;
        ob[(((size_t)(bb * Hn + blockIdx.x)) * HDc + d) * Tc + t] = f2bf(v);
      } else {                     // fp32 row-major [M, N]
        float* of = (float*)outp;
        of[(size_t)m * N + n0 + d] = v;
      }
    }
  }
}

// ---------------------------------------------------------------------------
// Flash attention.  Grid (T/128, H, B); 256 threads = 8 waves.
// Wave w owns q rows [w*16, w*16+16); Q fragments live in registers.
// KV tiles of 64 columns staged by the Tensor Data Mover (wave 0 issues the
// descriptor, waits TENSORcnt, then the workgroup barrier publishes the LDS).
// ---------------------------------------------------------------------------
#define TQ 128
#define TJ 64

__global__ __launch_bounds__(256) void attn_kernel(
    const __bf16* __restrict__ qb, const __bf16* __restrict__ kb,
    const __bf16* __restrict__ vb, __bf16* __restrict__ ob) {
  __shared__ __bf16 Kt[TJ][HDc];      // 16 KB : K rows [j][d]
  __shared__ __bf16 Vt[HDc][TJ];      // 16 KB : V^T   [d][j]
  __shared__ __bf16 Pt[8][16][TJ];    // 16 KB : per-wave P scratch

  const int b = blockIdx.z, h = blockIdx.y, qt = blockIdx.x;
  const int kvh = h >> 3;                         // 16/2 = 8-way GQA
  const int wave = threadIdx.x >> 5;
  const int lane = threadIdx.x & 31;
  const int half = lane >> 4;
  const int ncol = lane & 15;

  // Q fragments (4 K-chunks of 32 over HEAD_DIM=128), straight from global.
  const __bf16* qrow =
      qb + (((size_t)(b * Hc + h)) * Tc + qt * TQ + wave * 16) * HDc;
  v16bf qf[4];
#pragma unroll
  for (int c = 0; c < 4; ++c) qf[c] = ld_frag(qrow, HDc, c * 32, lane);

  const v8f vzero = {0.f, 0.f, 0.f, 0.f, 0.f, 0.f, 0.f, 0.f};
  v8f oacc[8];
#pragma unroll
  for (int i = 0; i < 8; ++i) oacc[i] = vzero;
  float mrow[8], lrow[8];
#pragma unroll
  for (int r = 0; r < 8; ++r) { mrow[r] = -1e30f; lrow[r] = 0.f; }

  const float scale = 0.08838834764831845f;       // 1/sqrt(128)
  const int njt = qt * (TQ / TJ) + (TQ / TJ);     // causal tile count

  // K slab [T][HD] and transposed V slab [HD][T] for this (b, kvh).
  const __bf16* kslab = kb + ((size_t)(b * KVHc + kvh)) * Tc * HDc;
  const __bf16* vslab = vb + ((size_t)(b * KVHc + kvh)) * HDc * Tc;

  for (int jt = 0; jt < njt; ++jt) {
    __syncthreads();                        // previous iteration's reads done
    if (wave == 0) {
      // K tile: tensor [T][128], tile = rows jt*TJ..+63 (contiguous) -> Kt
      tdm_load_2d(kslab + (size_t)(jt * TJ) * HDc, lds_off(&Kt[0][0]),
                  /*tensor_d0=*/HDc, /*tensor_d1=*/Tc,
                  /*tile_d0=*/HDc, /*tile_d1=*/TJ, /*stride0=*/HDc);
      // V tile: tensor [128][T], tile = cols jt*TJ..+63 of all 128 rows -> Vt
      tdm_load_2d(vslab + jt * TJ, lds_off(&Vt[0][0]),
                  /*tensor_d0=*/Tc, /*tensor_d1=*/HDc,
                  /*tile_d0=*/TJ, /*tile_d1=*/HDc, /*stride0=*/Tc);
      __builtin_amdgcn_s_wait_tensorcnt(0);
    }
    if (jt + 1 < njt) {                     // gfx1250 global_prefetch_b8
      __builtin_prefetch(kslab + (size_t)((jt + 1) * TJ) * HDc + threadIdx.x * 32,
                         0, 1);
      __builtin_prefetch(vslab + (jt + 1) * TJ + (size_t)threadIdx.x * Tc, 0, 1);
    }
    __syncthreads();

    // S = Q K^T  (16 x TJ per wave; B operand taken from Kt[j][d] directly).
    v8f sacc[4];
#pragma unroll
    for (int i = 0; i < 4; ++i) sacc[i] = vzero;
#pragma unroll
    for (int c = 0; c < 4; ++c) {
      v16bf a = qf[c];
#pragma unroll
      for (int ni = 0; ni < 4; ++ni) {
        v16bf bf = ld_frag(&Kt[ni * 16][0], HDc, c * 32, lane);
        sacc[ni] = wmma_bf16(a, bf, sacc[ni]);
      }
    }

    // Causal mask + scale + online softmax (row reductions within 16 lanes;
    // xor masks < 16 keep the two C-layout row halves independent).
    const int ig_base = qt * TQ + wave * 16 + (half << 3);
    float mnew[8];
#pragma unroll
    for (int r = 0; r < 8; ++r) {
      int ig = ig_base + r;
      float mx = -1e30f;
#pragma unroll
      for (int ni = 0; ni < 4; ++ni) {
        int jg = jt * TJ + (ni << 4) + ncol;
        float s = sacc[ni][r] * scale;
        s = (jg <= ig) ? s : -1e9f;
        sacc[ni][r] = s;
        mx = fmaxf(mx, s);
      }
#pragma unroll
      for (int off = 1; off < 16; off <<= 1)
        mx = fmaxf(mx, __shfl_xor(mx, off, 32));
      mnew[r] = fmaxf(mrow[r], mx);
    }
#pragma unroll
    for (int r = 0; r < 8; ++r) {
      float corr = __expf(mrow[r] - mnew[r]);
      mrow[r] = mnew[r];
      float rs = 0.f;
#pragma unroll
      for (int ni = 0; ni < 4; ++ni) {
        float p = __expf(sacc[ni][r] - mnew[r]);
        sacc[ni][r] = p;
        rs += p;
      }
#pragma unroll
      for (int off = 1; off < 16; off <<= 1) rs += __shfl_xor(rs, off, 32);
      lrow[r] = lrow[r] * corr + rs;
#pragma unroll
      for (int ni2 = 0; ni2 < 8; ++ni2) oacc[ni2][r] *= corr;
    }

    // Re-fragment P through per-wave LDS scratch (C-layout -> A-layout).
#pragma unroll
    for (int ni = 0; ni < 4; ++ni)
#pragma unroll
      for (int r = 0; r < 8; ++r)
        Pt[wave][r + (half << 3)][(ni << 4) + ncol] = f2bf(sacc[ni][r]);
    __syncthreads();

    // O += P * V   (A = Pt[wave] 16 x TJ, B from Vt[d][j]).
#pragma unroll
    for (int c2 = 0; c2 < TJ / 32; ++c2) {
      v16bf a = ld_frag(&Pt[wave][0][0], TJ, c2 * 32, lane);
#pragma unroll
      for (int ni2 = 0; ni2 < 8; ++ni2) {
        v16bf bf = ld_frag(&Vt[ni2 * 16][0], TJ, c2 * 32, lane);
        oacc[ni2] = wmma_bf16(a, bf, oacc[ni2]);
      }
    }
  }

  // Normalize and store to attention output [B, T, H*HD] (bf16 GEMM input).
#pragma unroll
  for (int ni2 = 0; ni2 < 8; ++ni2)
#pragma unroll
    for (int r = 0; r < 8; ++r) {
      int t = qt * TQ + wave * 16 + (half << 3) + r;
      int d = (ni2 << 4) + ncol;
      float v = oacc[ni2][r] / lrow[r];
      ob[((size_t)b * Tc + t) * DIMc + h * HDc + d] = f2bf(v);
    }
}

// ---------------------------------------------------------------------------
// Host launcher.
// ---------------------------------------------------------------------------
extern "C" void kernel_launch(void* const* d_in, const int* in_sizes, int n_in,
                              void* d_out, int out_size, void* d_ws,
                              size_t ws_size, hipStream_t stream) {
  const float* x    = (const float*)d_in[0];
  const float* cosb = (const float*)d_in[1];
  const float* sinb = (const float*)d_in[2];
  const float* wq   = (const float*)d_in[3];
  const float* wqb  = (const float*)d_in[4];
  const float* wk   = (const float*)d_in[5];
  const float* wkb  = (const float*)d_in[6];
  const float* wv   = (const float*)d_in[7];
  const float* wvb  = (const float*)d_in[8];
  const float* wo   = (const float*)d_in[9];
  float* out = (float*)d_out;

  char* p = (char*)d_ws;
  auto carve = [&](size_t bytes) -> char* {
    char* r = p;
    p += (bytes + 255) & ~(size_t)255;
    return r;
  };
  __bf16* xb  = (__bf16*)carve((size_t)Mc * Kc * 2);            // x bf16
  __bf16* wqT = (__bf16*)carve((size_t)DIMc * DIMc * 2);        // [N][K]
  __bf16* wkT = (__bf16*)carve((size_t)(KVHc * HDc) * DIMc * 2);
  __bf16* wvT = (__bf16*)carve((size_t)(KVHc * HDc) * DIMc * 2);
  __bf16* woT = (__bf16*)carve((size_t)DIMc * DIMc * 2);
  __bf16* qb  = (__bf16*)carve((size_t)Bc * Hc * Tc * HDc * 2);   // [B,H,T,HD]
  __bf16* kb  = (__bf16*)carve((size_t)Bc * KVHc * Tc * HDc * 2); // [B,KVH,T,HD]
  __bf16* vb  = (__bf16*)carve((size_t)Bc * KVHc * HDc * Tc * 2); // [B,KVH,HD,T]
  __bf16* ab  = (__bf16*)carve((size_t)Mc * DIMc * 2);            // [B,T,DIM]

  // 1) precision / layout conversion passes
  {
    int n = Mc * Kc;
    cvt_f32_bf16<<<2048, 256, 0, stream>>>(x, xb, n);
  }
  tr_f32_bf16<<<(DIMc * DIMc + 255) / 256, 256, 0, stream>>>(wq, wqT, DIMc, DIMc);
  tr_f32_bf16<<<(DIMc * KVHc * HDc + 255) / 256, 256, 0, stream>>>(wk, wkT, DIMc,
                                                                   KVHc * HDc);
  tr_f32_bf16<<<(DIMc * KVHc * HDc + 255) / 256, 256, 0, stream>>>(wv, wvT, DIMc,
                                                                   KVHc * HDc);
  tr_f32_bf16<<<(DIMc * DIMc + 255) / 256, 256, 0, stream>>>(wo, woT, DIMc, DIMc);

  // 2) fused QKV projections (+bias, +RoPE for Q/K; V stored transposed)
  gemm_kernel<0, true, true><<<dim3(DIMc / 128, Mc / 128), 256, 0, stream>>>(
      xb, wqT, wqb, cosb, sinb, qb, DIMc, Hc);
  gemm_kernel<0, true, true><<<dim3(KVHc * HDc / 128, Mc / 128), 256, 0, stream>>>(
      xb, wkT, wkb, cosb, sinb, kb, KVHc * HDc, KVHc);
  gemm_kernel<1, false, true><<<dim3(KVHc * HDc / 128, Mc / 128), 256, 0, stream>>>(
      xb, wvT, wvb, nullptr, nullptr, vb, KVHc * HDc, KVHc);

  // 3) flash attention (TDM-staged K/V)
  attn_kernel<<<dim3(Tc / TQ, Hc, Bc), 256, 0, stream>>>(qb, kb, vb, ab);

  // 4) output projection -> fp32 d_out
  gemm_kernel<2, false, false><<<dim3(DIMc / 128, Mc / 128), 256, 0, stream>>>(
      ab, woT, nullptr, nullptr, nullptr, out, DIMc, Hc);
}
